// nrecGNN_45019847197184
// MI455X (gfx1250) — compile-verified, matrix operand order checked by reading, not run on hardware
//
#include <hip/hip_runtime.h>
#include <hip/hip_bf16.h>

typedef __attribute__((ext_vector_type(16))) _Float16 v16h;
typedef __attribute__((ext_vector_type(8)))  _Float16 v8h;
typedef __attribute__((ext_vector_type(8)))  float    v8f;

typedef unsigned int tdm_u32x4 __attribute__((ext_vector_type(4)));
typedef int          tdm_i32x8 __attribute__((ext_vector_type(8)));
typedef int          tdm_i32x4 __attribute__((ext_vector_type(4)));

#if __has_builtin(__builtin_amdgcn_tensor_load_to_lds) && __has_builtin(__builtin_amdgcn_s_wait_tensorcnt)
#define USE_TDM 1
#endif

#define D128 128

__device__ __forceinline__ float wave_sum(float v) {
#pragma unroll
    for (int o = 16; o > 0; o >>= 1) v += __shfl_xor(v, o, 32);
    return v;
}
__device__ __forceinline__ float wave_max(float v) {
#pragma unroll
    for (int o = 16; o > 0; o >>= 1) v = fmaxf(v, __shfl_xor(v, o, 32));
    return v;
}

// ---------------------------------------------------------------------------
// prep: xh = f16(x);  anchorh = f16(l2norm(x))      one wave32 per node
// ---------------------------------------------------------------------------
__global__ void prep_kernel(const float* __restrict__ x,
                            _Float16* __restrict__ xh,
                            _Float16* __restrict__ anchorh, int Nn)
{
    int wid  = (int)((blockIdx.x * blockDim.x + threadIdx.x) >> 5);
    int lane = threadIdx.x & 31;
    if (wid >= Nn) return;
    const float4 v = ((const float4*)(x + (long)wid * D128))[lane];
    float ss  = wave_sum(v.x * v.x + v.y * v.y + v.z * v.z + v.w * v.w);
    float inv = 1.0f / fmaxf(sqrtf(ss), 1e-12f);
    union { _Float16 h[4]; uint2 u; } p;
    p.h[0] = (_Float16)v.x; p.h[1] = (_Float16)v.y;
    p.h[2] = (_Float16)v.z; p.h[3] = (_Float16)v.w;
    ((uint2*)(xh + (long)wid * D128))[lane] = p.u;
    p.h[0] = (_Float16)(v.x * inv); p.h[1] = (_Float16)(v.y * inv);
    p.h[2] = (_Float16)(v.z * inv); p.h[3] = (_Float16)(v.w * inv);
    ((uint2*)(anchorh + (long)wid * D128))[lane] = p.u;
}

// ---------------------------------------------------------------------------
// pack weight [K=128, Nw] (row-major f32) into WMMA B-operand lane-major f16:
// Bp[((ks*NT+nt)*32 + lane)*16 + i]  where lane holds col n = nt*16+(lane&15)
// and element i maps to K = ks*32 + (i<8 ? hl*8+i : 16+hl*8+(i-8)), hl=lane>>4
// ---------------------------------------------------------------------------
__global__ void pack_b_kernel(const float* __restrict__ W,
                              _Float16* __restrict__ Bp, int Nw, int NT)
{
    int gid = blockIdx.x * blockDim.x + threadIdx.x;
    int total = 4 * NT * 512;
    if (gid >= total) return;
    int i    = gid & 15;
    int lane = (gid >> 4) & 31;
    int t    = gid >> 9;
    int nt   = t % NT;
    int ks   = t / NT;
    int hl   = lane >> 4;
    int n    = nt * 16 + (lane & 15);
    int k    = ks * 32 + (i < 8 ? hl * 8 + i : 16 + hl * 8 + (i - 8));
    float v  = (n < Nw) ? W[k * Nw + n] : 0.0f;
    Bp[gid]  = (_Float16)v;
}

__global__ void padbias_kernel(const float* __restrict__ b,
                               float* __restrict__ bp, int n, int npad)
{
    int i = threadIdx.x;
    if (i < npad) bp[i] = (i < n) ? b[i] : 0.0f;
}

__global__ void zero_kernel(float4* __restrict__ p, long n4)
{
    long i = (long)blockIdx.x * blockDim.x + threadIdx.x;
    if (i < n4) { float4 z; z.x = z.y = z.z = z.w = 0.0f; p[i] = z; }
}

// ---------------------------------------------------------------------------
// WMMA GEMM: [M x 128] f16 @ [128 x NT*16].  One wave per 16-row stripe.
// B staged to LDS once per block via the Tensor Data Mover (async, TENSORcnt),
// inner loop reads B with ds_load and feeds v_wmma_f32_16x16x32_f16.
// MODE 0: store f32.  MODE 1: +bias, ReLU, store f16.  MODE 2: +bias, f32.
// ---------------------------------------------------------------------------
template <int MODE, int NT>
__global__ void gemm_wmma_kernel(const _Float16* __restrict__ A,
                                 const _Float16* __restrict__ Bp,
                                 const float* __restrict__ bias,
                                 void* __restrict__ Cout, int M, int ldc)
{
    const int lane  = threadIdx.x & 31;
    const int mtile = (int)((blockIdx.x * blockDim.x + threadIdx.x) >> 5);
    const int hl    = lane >> 4;
    const int mrow  = lane & 15;

    __shared__ _Float16 Bs[4 * NT * 512];

#ifdef USE_TDM
    if (threadIdx.x < 32) {            // wave 0 issues one TDM copy (EXEC ignored)
        const unsigned nelem = 4u * NT * 512u;         // f16 elements (tile_dim0)
        unsigned long long ga = (unsigned long long)(uintptr_t)Bp;
        unsigned lds = (unsigned)(uintptr_t)(void*)Bs;
        tdm_u32x4 g0;
        g0.x = 1u;                                      // count=1, user mode
        g0.y = lds;                                     // lds_addr
        g0.z = (unsigned)(ga & 0xFFFFFFFFu);            // global_addr[31:0]
        g0.w = (unsigned)((ga >> 32) & 0x1FFFFFFu) | (2u << 30);  // addr[56:32]|type=2
        tdm_i32x8 g1;
        g1[0] = (int)(1u << 16);                        // data_size=2B; mask=0
        g1[1] = (int)((nelem & 0xFFFFu) << 16);         // tensor_dim0 lo16
        g1[2] = (int)((nelem >> 16) & 0xFFFFu) | (1 << 16); // dim0 hi | tensor_dim1=1
        g1[3] = (int)((nelem & 0xFFFFu) << 16);         // tile_dim0 = nelem
        g1[4] = 1;                                      // tile_dim1=1
        g1[5] = (int)nelem;                             // tensor_dim0_stride lo32
        g1[6] = 0;
        g1[7] = 0;
        tdm_i32x4 gz4 = {0, 0, 0, 0};
        tdm_i32x8 gz8 = {0, 0, 0, 0, 0, 0, 0, 0};
        __builtin_amdgcn_tensor_load_to_lds(g0, g1, gz4, gz4, gz8, 0);
        __builtin_amdgcn_s_wait_tensorcnt(0);
    }
#else
    for (int t = threadIdx.x; t < NT * 256; t += blockDim.x)
        ((uint4*)Bs)[t] = ((const uint4*)Bp)[t];
#endif
    __syncthreads();

    if (mtile * 16 >= M) return;       // uniform per wave; after the barrier

    const _Float16* arow = A + ((long)mtile * 16 + mrow) * D128;
    v16h a[4];
#pragma unroll
    for (int ks = 0; ks < 4; ++ks) {
        union { v16h v; v8h h[2]; } av;
        av.h[0] = *(const v8h*)(arow + ks * 32 + hl * 8);
        av.h[1] = *(const v8h*)(arow + ks * 32 + 16 + hl * 8);
        a[ks] = av.v;
    }

    v8f acc[NT] = {};
#pragma unroll
    for (int ks = 0; ks < 4; ++ks) {
#pragma unroll
        for (int nt = 0; nt < NT; ++nt) {
            v16h b = *(const v16h*)(Bs + ((ks * NT + nt) * 32 + lane) * 16);
            acc[nt] = __builtin_amdgcn_wmma_f32_16x16x32_f16(
                false, a[ks], false, b, (short)0, acc[nt], false, false);
        }
    }

#pragma unroll
    for (int nt = 0; nt < NT; ++nt) {
        const int col = nt * 16 + mrow;
        const float bv = (MODE >= 1) ? bias[col] : 0.0f;
#pragma unroll
        for (int r = 0; r < 8; ++r) {
            long orow = (long)mtile * 16 + hl * 8 + r;
            float v = acc[nt][r] + bv;
            if (MODE == 1) {
                v = fmaxf(v, 0.0f);
                ((_Float16*)Cout)[orow * ldc + col] = (_Float16)v;
            } else {
                ((float*)Cout)[orow * ldc + col] = v;
            }
        }
    }
}

// ---------------------------------------------------------------------------
// SpMM scatter: hop[h][r] += v * xw[c]    one wave per edge, float4 per lane
// xw (25.6 MB) is L2-resident (192 MB), so the 3x256 MB gather stays on-chip.
// ---------------------------------------------------------------------------
__global__ void spmm_kernel(const int* __restrict__ rows,
                            const int* __restrict__ cols,
                            const float* __restrict__ vals,
                            const float* __restrict__ xw,
                            float* __restrict__ hop, int nnz, int Nn)
{
    long gid = (long)blockIdx.x * blockDim.x + threadIdx.x;
    long e   = gid >> 5;
    if (e >= nnz) return;
    int h    = blockIdx.y;
    int lane = threadIdx.x & 31;
    long eo  = (long)h * nnz + e;
    int r    = rows[eo];
    int c    = cols[eo];
    float v  = vals[eo];
    float4 m = ((const float4*)(xw + (long)c * D128))[lane];
    float* dst = hop + ((long)h * Nn + r) * D128 + lane * 4;
    atomicAdd(dst + 0, m.x * v);
    atomicAdd(dst + 1, m.y * v);
    atomicAdd(dst + 2, m.z * v);
    atomicAdd(dst + 3, m.w * v);
}

__global__ void hopnorm_kernel(float* __restrict__ hop, long nrows)
{
    long wid = ((long)blockIdx.x * blockDim.x + threadIdx.x) >> 5;
    int lane = threadIdx.x & 31;
    if (wid >= nrows) return;
    float4* p = (float4*)(hop + wid * D128);
    float4 v = p[lane];
    float ss  = wave_sum(v.x * v.x + v.y * v.y + v.z * v.z + v.w * v.w);
    float inv = 1.0f / fmaxf(sqrtf(ss), 1e-12f);
    v.x *= inv; v.y *= inv; v.z *= inv; v.w *= inv;
    p[lane] = v;
}

// ---------------------------------------------------------------------------
// attention over {anchor, hop0, hop1, hop2}: one wave per node
// ---------------------------------------------------------------------------
__global__ void attn_kernel(const float* __restrict__ x,
                            const float* __restrict__ q,
                            const float* __restrict__ hops,
                            _Float16* __restrict__ outh, int Nn)
{
    int wid  = (int)((blockIdx.x * blockDim.x + threadIdx.x) >> 5);
    int lane = threadIdx.x & 31;
    if (wid >= Nn) return;
    float4 xa = ((const float4*)(x + (long)wid * D128))[lane];
    float ssa = wave_sum(xa.x * xa.x + xa.y * xa.y + xa.z * xa.z + xa.w * xa.w);
    float inv = 1.0f / fmaxf(sqrtf(ssa), 1e-12f);
    float4 a;  a.x = xa.x * inv; a.y = xa.y * inv; a.z = xa.z * inv; a.w = xa.w * inv;
    float4 qv = ((const float4*)(q + (long)wid * D128))[lane];
    float4 h0 = ((const float4*)(hops + ((long)0 * Nn + wid) * D128))[lane];
    float4 h1 = ((const float4*)(hops + ((long)1 * Nn + wid) * D128))[lane];
    float4 h2 = ((const float4*)(hops + ((long)2 * Nn + wid) * D128))[lane];

    float s0 = wave_sum(qv.x * a.x + qv.y * a.y + qv.z * a.z + qv.w * a.w);
    float s1 = wave_sum(qv.x * h0.x + qv.y * h0.y + qv.z * h0.z + qv.w * h0.w);
    float s2 = wave_sum(qv.x * h1.x + qv.y * h1.y + qv.z * h1.z + qv.w * h1.w);
    float s3 = wave_sum(qv.x * h2.x + qv.y * h2.y + qv.z * h2.z + qv.w * h2.w);
    const float sc = 0.0883883476483184f;  // 1/sqrt(128)
    s0 *= sc; s1 *= sc; s2 *= sc; s3 *= sc;
    float mx = fmaxf(fmaxf(s0, s1), fmaxf(s2, s3));
    float e0 = expf(s0 - mx), e1 = expf(s1 - mx), e2 = expf(s2 - mx), e3 = expf(s3 - mx);
    float rs = 1.0f / (e0 + e1 + e2 + e3);
    float w0 = e0 * rs, w1 = e1 * rs, w2 = e2 * rs, w3 = e3 * rs;

    union { _Float16 h[4]; uint2 u; } p;
    p.h[0] = (_Float16)(a.x * w0 + h0.x * w1 + h1.x * w2 + h2.x * w3);
    p.h[1] = (_Float16)(a.y * w0 + h0.y * w1 + h1.y * w2 + h2.y * w3);
    p.h[2] = (_Float16)(a.z * w0 + h0.z * w1 + h1.z * w2 + h2.z * w3);
    p.h[3] = (_Float16)(a.w * w0 + h0.w * w1 + h1.w * w2 + h2.w * w3);
    ((uint2*)(outh + (long)wid * D128))[lane] = p.u;
}

// ---------------------------------------------------------------------------
// log_softmax over 40 classes (logits row stride 48): one wave per node
// ---------------------------------------------------------------------------
__global__ void lsm_kernel(const float* __restrict__ logits,
                           float* __restrict__ out, int Nn)
{
    int wid  = (int)((blockIdx.x * blockDim.x + threadIdx.x) >> 5);
    int lane = threadIdx.x & 31;
    if (wid >= Nn) return;
    const float* lr = logits + (long)wid * 48;
    const float NEG = -3.402823466e38f;
    float l0 = (lane < 40) ? lr[lane] : NEG;
    float l1 = (lane + 32 < 40) ? lr[lane + 32] : NEG;
    float m  = wave_max(fmaxf(l0, l1));
    float s  = wave_sum(((lane < 40) ? expf(l0 - m) : 0.0f) +
                        ((lane + 32 < 40) ? expf(l1 - m) : 0.0f));
    float lse = m + logf(s);
    if (lane < 40)      out[(long)wid * 40 + lane]      = l0 - lse;
    if (lane + 32 < 40) out[(long)wid * 40 + lane + 32] = l1 - lse;
}

// ---------------------------------------------------------------------------
extern "C" void kernel_launch(void* const* d_in, const int* in_sizes, int n_in,
                              void* d_out, int out_size, void* d_ws, size_t ws_size,
                              hipStream_t stream)
{
    const float* x    = (const float*)d_in[0];
    const int*   rows = (const int*)d_in[2];
    const int*   cols = (const int*)d_in[3];
    const float* vals = (const float*)d_in[4];
    const float* Wg   = (const float*)d_in[5];
    const float* Wq   = (const float*)d_in[6];
    const float* W1   = (const float*)d_in[7];
    const float* b1   = (const float*)d_in[8];
    const float* W2   = (const float*)d_in[9];
    const float* b2   = (const float*)d_in[10];
    const int Nn  = in_sizes[0] / D128;
    const int nnz = in_sizes[2] / 3;
    float* out = (float*)d_out;

    char* ws = (char*)d_ws;
    size_t off = 0;
    auto alloc = [&](size_t bytes) -> void* {
        void* p = ws + off;
        off += (bytes + 255) & ~(size_t)255;
        return p;
    };
    _Float16* xh    = (_Float16*)alloc((size_t)Nn * D128 * 2);      // -> hrelu later
    _Float16* anchh = (_Float16*)alloc((size_t)Nn * D128 * 2);      // -> outh later
    float*    xw    = (float*)alloc((size_t)Nn * D128 * 4);         // -> logits later
    float*    q     = (float*)alloc((size_t)Nn * D128 * 4);
    float*    hops  = (float*)alloc((size_t)3 * Nn * D128 * 4);
    _Float16* WgP   = (_Float16*)alloc(4 * 8 * 512 * 2);
    _Float16* WqP   = (_Float16*)alloc(4 * 8 * 512 * 2);
    _Float16* W1P   = (_Float16*)alloc(4 * 8 * 512 * 2);
    _Float16* W2P   = (_Float16*)alloc(4 * 3 * 512 * 2);
    float*    b2p   = (float*)alloc(48 * 4);

    const dim3 blk(256);
    // stage 0: prep + weight packing
    prep_kernel<<<(Nn * 32 + 255) / 256, blk, 0, stream>>>(x, xh, anchh, Nn);
    pack_b_kernel<<<(4 * 8 * 512 + 255) / 256, blk, 0, stream>>>(Wg, WgP, 128, 8);
    pack_b_kernel<<<(4 * 8 * 512 + 255) / 256, blk, 0, stream>>>(Wq, WqP, 128, 8);
    pack_b_kernel<<<(4 * 8 * 512 + 255) / 256, blk, 0, stream>>>(W1, W1P, 128, 8);
    pack_b_kernel<<<(4 * 3 * 512 + 255) / 256, blk, 0, stream>>>(W2, W2P, 40, 3);
    padbias_kernel<<<1, 64, 0, stream>>>(b2, b2p, 40, 48);
    long n4 = (long)3 * Nn * 32;
    zero_kernel<<<(unsigned)((n4 + 255) / 256), blk, 0, stream>>>((float4*)hops, n4);

    const int mtiles  = (Nn + 15) / 16;
    const int gblocks = (mtiles + 7) / 8;  // 8 waves / block

    // stage 1: xw = x @ Wg  (WMMA, TDM-staged B)
    gemm_wmma_kernel<0, 8><<<gblocks, blk, 0, stream>>>(xh, WgP, nullptr, xw, Nn, 128);

    // stage 2: hops[h] = segment_sum(vals * xw[cols], rows); then per-row l2norm
    dim3 sgrid((unsigned)(((long)nnz * 32 + 255) / 256), 3);
    spmm_kernel<<<sgrid, blk, 0, stream>>>(rows, cols, vals, xw, hops, nnz, Nn);
    long hw = (long)3 * Nn;
    hopnorm_kernel<<<(unsigned)((hw * 32 + 255) / 256), blk, 0, stream>>>(hops, hw);

    // stage 3: q = anchor @ Wq  (WMMA)
    gemm_wmma_kernel<0, 8><<<gblocks, blk, 0, stream>>>(anchh, WqP, nullptr, q, Nn, 128);

    // stage 4: 4-way softmax attention -> outh (f16), reuse anchh storage
    _Float16* outh = anchh;
    attn_kernel<<<(Nn * 32 + 255) / 256, blk, 0, stream>>>(x, q, hops, outh, Nn);

    // stage 5: hrelu = relu(out @ W1 + b1)  (WMMA, f16 out), reuse xh storage
    _Float16* hrelu = xh;
    gemm_wmma_kernel<1, 8><<<gblocks, blk, 0, stream>>>(outh, W1P, b1, hrelu, Nn, 128);

    // stage 6: logits = hrelu @ W2 + b2  (WMMA, padded N=48), reuse xw storage
    float* logits = xw;
    gemm_wmma_kernel<2, 3><<<gblocks, blk, 0, stream>>>(hrelu, W2P, b2p, logits, Nn, 48);

    // stage 7: log_softmax over 40 classes
    lsm_kernel<<<(Nn * 32 + 255) / 256, blk, 0, stream>>>(logits, out, Nn);
}
